// RelationalAttLayer_1760936591782
// MI455X (gfx1250) — matrix operand order repaired and checked by reading, not compiled, not run on hardware
//
#include <hip/hip_runtime.h>
#include <hip/hip_bf16.h>

// ---------------------------------------------------------------------------
// RelationalAttLayer (R-GAT, R=3, N=50000, E=800000, IN=OUT=128, H=4, D=32)
// Strategy:
//   - x@W_r and x@loop_weight via v_wmma_f32_16x16x32_f16 (f32 accum).
//   - A/B operands pre-packed as WMMA fragments with contiguous k (k = 16*(lane/16) + i)
//     so the GEMM does contiguous 32B/lane loads (A from global, B from LDS).
//   - Edge softmax: 3 streaming passes with f32 atomics (punned float max).
//   - Self-loop GEMM writes acc directly (no zero pass); edge aggregation
//     atomically adds on top; final kernel adds bias + relu.
// ---------------------------------------------------------------------------

#define NN      50000
#define EE      800000
#define RR      3
#define INF_    128
#define OUTF    128
#define HH      4
#define DD      32
#define NTILES  3125          // NN / 16
#define NEGSL   0.2f

typedef __attribute__((ext_vector_type(16))) _Float16 v16h;
typedef __attribute__((ext_vector_type(8)))  float    v8f;

// ---------------- pack x into f16 WMMA A-fragments ----------------
// layout: [tile(3125)][kt(4)][lane(32)][i(16)] halves
// lane<16 : row = tile*16+lane,    K = i                        (0..15)
// lane>=16: row = tile*16+lane-16, K = i + 16                   (16..31)
__global__ void pack_x_kernel(const float* __restrict__ x, _Float16* __restrict__ xpack) {
    int idx = blockIdx.x * 256 + threadIdx.x;           // 6,400,000 total
    if (idx >= NTILES * 4 * 32 * 16) return;
    int i    = idx & 15;
    int lane = (idx >> 4) & 31;
    int kt   = (idx >> 9) & 3;
    int tile = idx >> 11;
    int row  = tile * 16 + (lane & 15);
    int kk   = kt * 32 + i + ((lane >> 4) ? 16 : 0);
    xpack[idx] = (_Float16)x[(size_t)row * INF_ + kk];
}

// ---------------- pack W (3 rels + loop) into f16 WMMA B-fragments ----------
// layout: [m(4)][kt(4)][j(8)][lane(32)][i(16)] halves
// lane  -> column j*16 + (lane&15); K = kt*32 + i + (lane>=16 ? 16 : 0)
__global__ void pack_w_kernel(const float* __restrict__ W, const float* __restrict__ loopw,
                              _Float16* __restrict__ wpack) {
    int idx = blockIdx.x * 256 + threadIdx.x;           // 65,536 total
    if (idx >= 4 * 4 * 8 * 32 * 16) return;
    int i    = idx & 15;
    int lane = (idx >> 4) & 31;
    int j    = (idx >> 9) & 7;
    int kt   = (idx >> 12) & 3;
    int m    = idx >> 14;
    int col  = j * 16 + (lane & 15);
    int kk   = kt * 32 + i + ((lane >> 4) ? 16 : 0);
    float v = (m < 3) ? W[((size_t)m * INF_ + kk) * OUTF + col]
                      : loopw[(size_t)kk * OUTF + col];
    wpack[idx] = (_Float16)v;
}

// ---------------- WMMA GEMM: out[N,128] = x @ Wm  ----------------
// 8 waves/block, one 16-row strip per wave, 8 accum tiles (16x128 out),
// B matrix (32KB packed f16) staged in LDS, 32 wmma per wave.
__global__ void gemm_wmma_kernel(const _Float16* __restrict__ xpack,
                                 const _Float16* __restrict__ wpack_m,
                                 float* __restrict__ out) {
    __shared__ __align__(32) _Float16 sB[16384];        // 32 KB: [kt][j][lane][i]
    {
        const uint4* s = reinterpret_cast<const uint4*>(wpack_m);
        uint4*       d = reinterpret_cast<uint4*>(sB);
        #pragma unroll
        for (int t = 0; t < 8; ++t) d[threadIdx.x + t * 256] = s[threadIdx.x + t * 256];
    }
    __syncthreads();

    int wave = threadIdx.x >> 5;
    int lane = threadIdx.x & 31;
    int tile = blockIdx.x * 8 + wave;
    if (tile >= NTILES) return;

    v8f acc[8] = {};
    const v16h* ap = reinterpret_cast<const v16h*>(xpack) + (size_t)tile * 4 * 32 + lane;
    const v16h* bp = reinterpret_cast<const v16h*>(sB) + lane;

    #pragma unroll
    for (int kt = 0; kt < 4; ++kt) {
        v16h a = ap[kt * 32];
        #pragma unroll
        for (int j = 0; j < 8; ++j) {
            v16h b = bp[(kt * 8 + j) * 32];
            acc[j] = __builtin_amdgcn_wmma_f32_16x16x32_f16(
                false, a, false, b, (short)0, acc[j], false, false);
        }
    }

    // C/D layout: lane&15 = column, VGPR v = row (lane>=16 -> rows 8..15)
    int m0   = tile * 16 + ((lane >> 4) ? 8 : 0);
    int col0 = lane & 15;
    #pragma unroll
    for (int j = 0; j < 8; ++j) {
        #pragma unroll
        for (int v = 0; v < 8; ++v) {
            out[(size_t)(m0 + v) * OUTF + j * 16 + col0] = acc[j][v];
        }
    }
}

// ---------------- per-node attention dots: el/er [N,H] ----------------
__global__ void attn_dots_kernel(const float* __restrict__ h,
                                 const float* __restrict__ al, const float* __restrict__ ar,
                                 float* __restrict__ el, float* __restrict__ er) {
    int idx = blockIdx.x * 256 + threadIdx.x;           // N*H = 200,000
    if (idx >= NN * HH) return;
    int hh = idx & 3, n = idx >> 2;
    const float* hp  = h + (size_t)n * OUTF + hh * DD;
    const float* alp = al + hh * DD;
    const float* arp = ar + hh * DD;
    float sl = 0.f, sr = 0.f;
    #pragma unroll
    for (int d = 0; d < DD; ++d) { float v = hp[d]; sl += v * alp[d]; sr += v * arp[d]; }
    el[idx] = sl; er[idx] = sr;
}

__global__ void init_ms_kernel(float* __restrict__ m, float* __restrict__ s) {
    int idx = blockIdx.x * 256 + threadIdx.x;
    if (idx >= NN * HH) return;
    m[idx] = -__builtin_huge_valf();
    s[idx] = 0.f;
}

// punned float atomic max (handles mixed signs; init must be -inf)
__device__ __forceinline__ void atomicMaxFloat(float* addr, float val) {
    if (val >= 0.f) atomicMax((int*)addr, __float_as_int(val));
    else            atomicMin((unsigned int*)addr, __float_as_uint(val));
}

// ---------------- pass A: logits + segment max ----------------
__global__ void edge_logits_kernel(const int* __restrict__ src, const int* __restrict__ dst,
                                   const float* __restrict__ el, const float* __restrict__ er,
                                   float* __restrict__ ebuf, float* __restrict__ m) {
    int e = blockIdx.x * 256 + threadIdx.x;
    if (e >= EE) return;
    int s_ = src[e], d_ = dst[e];
    #pragma unroll
    for (int hh = 0; hh < HH; ++hh) {
        float v = el[s_ * HH + hh] + er[d_ * HH + hh];
        v = v > 0.f ? v : NEGSL * v;                    // leaky_relu
        ebuf[e * HH + hh] = v;
        atomicMaxFloat(&m[d_ * HH + hh], v);
    }
}

// ---------------- pass B: exp(e - m[dst]) + segment sum ----------------
__global__ void edge_exp_kernel(const int* __restrict__ dst, const float* __restrict__ m,
                                float* __restrict__ ebuf, float* __restrict__ s) {
    int e = blockIdx.x * 256 + threadIdx.x;
    if (e >= EE) return;
    int d_ = dst[e];
    #pragma unroll
    for (int hh = 0; hh < HH; ++hh) {
        float ex = __expf(ebuf[e * HH + hh] - m[d_ * HH + hh]);
        ebuf[e * HH + hh] = ex;
        atomicAdd(&s[d_ * HH + hh], ex);
    }
}

// ---------------- pass C: acc[dst] += alpha * h[src] ----------------
__global__ void edge_aggr_kernel(const int* __restrict__ src, const int* __restrict__ dst,
                                 const float* __restrict__ ebuf, const float* __restrict__ ssum,
                                 const float* __restrict__ h, float* __restrict__ acc) {
    long long idx = (long long)blockIdx.x * 256 + threadIdx.x;  // E*128 = 102,400,000
    if (idx >= (long long)EE * OUTF) return;
    int f  = (int)(idx & 127);
    int e  = (int)(idx >> 7);
    int s_ = src[e], d_ = dst[e];
    int hh = f >> 5;
    float alpha = ebuf[e * HH + hh] / ssum[d_ * HH + hh];
    atomicAdd(&acc[(size_t)d_ * OUTF + f], alpha * h[(size_t)s_ * OUTF + f]);
}

// ---------------- finalize: relu(acc + bias) ----------------
__global__ void finalize_kernel(const float* __restrict__ acc, const float* __restrict__ bias,
                                float* __restrict__ out) {
    int idx = blockIdx.x * 256 + threadIdx.x;           // 6,400,000
    if (idx >= NN * OUTF) return;
    float v = acc[idx] + bias[idx & (OUTF - 1)];
    out[idx] = v > 0.f ? v : 0.f;
}

extern "C" void kernel_launch(void* const* d_in, const int* in_sizes, int n_in,
                              void* d_out, int out_size, void* d_ws, size_t ws_size,
                              hipStream_t stream) {
    const float* x        = (const float*)d_in[0];      // [N,128]
    const float* W        = (const float*)d_in[1];      // [3,128,128]
    const float* attn_l   = (const float*)d_in[2];      // [3,4,32]
    const float* attn_r   = (const float*)d_in[3];      // [3,4,32]
    const float* loopw    = (const float*)d_in[4];      // [128,128]
    const float* h_bias   = (const float*)d_in[5];      // [128]
    const int*   edge_src = (const int*)d_in[6];        // [3,E]
    const int*   edge_dst = (const int*)d_in[7];        // [3,E]
    float*       out      = (float*)d_out;              // [N,128]

    // workspace carve-up (~80 MB, all chunks 256B-aligned)
    char* p = (char*)d_ws;
    _Float16* xpack = (_Float16*)p; p += (size_t)NN * INF_ * sizeof(_Float16);     // 12.8 MB
    _Float16* wpack = (_Float16*)p; p += (size_t)4 * INF_ * OUTF * sizeof(_Float16); // 128 KB
    float* hbuf = (float*)p; p += (size_t)NN * OUTF * sizeof(float);               // 25.6 MB
    float* acc  = (float*)p; p += (size_t)NN * OUTF * sizeof(float);               // 25.6 MB
    float* el   = (float*)p; p += (size_t)NN * HH * sizeof(float);                 // 0.8 MB
    float* er   = (float*)p; p += (size_t)NN * HH * sizeof(float);
    float* mbuf = (float*)p; p += (size_t)NN * HH * sizeof(float);
    float* sbuf = (float*)p; p += (size_t)NN * HH * sizeof(float);
    float* ebuf = (float*)p; p += (size_t)EE * HH * sizeof(float);                 // 12.8 MB

    // 1) pack operands into WMMA fragment layouts
    pack_x_kernel<<<25000, 256, 0, stream>>>(x, xpack);
    pack_w_kernel<<<256,   256, 0, stream>>>(W, loopw, wpack);

    // 2) self-loop GEMM writes acc directly (acts as accumulator init)
    gemm_wmma_kernel<<<(NTILES + 7) / 8, 256, 0, stream>>>(xpack, wpack + (size_t)3 * 16384, acc);

    // 3) per-relation GAT
    for (int r = 0; r < RR; ++r) {
        gemm_wmma_kernel<<<(NTILES + 7) / 8, 256, 0, stream>>>(
            xpack, wpack + (size_t)r * 16384, hbuf);
        attn_dots_kernel<<<(NN * HH + 255) / 256, 256, 0, stream>>>(
            hbuf, attn_l + r * HH * DD, attn_r + r * HH * DD, el, er);
        init_ms_kernel<<<(NN * HH + 255) / 256, 256, 0, stream>>>(mbuf, sbuf);
        edge_logits_kernel<<<EE / 256, 256, 0, stream>>>(
            edge_src + (size_t)r * EE, edge_dst + (size_t)r * EE, el, er, ebuf, mbuf);
        edge_exp_kernel<<<EE / 256, 256, 0, stream>>>(
            edge_dst + (size_t)r * EE, mbuf, ebuf, sbuf);
        edge_aggr_kernel<<<(int)(((long long)EE * OUTF) / 256), 256, 0, stream>>>(
            edge_src + (size_t)r * EE, edge_dst + (size_t)r * EE, ebuf, sbuf, hbuf, acc);
    }

    // 4) bias + relu
    finalize_kernel<<<(NN * OUTF + 255) / 256, 256, 0, stream>>>(acc, h_bias, out);
}
